// LocalRefinedAttention_32272384262295
// MI455X (gfx1250) — matrix operand order repaired and checked by reading, not compiled, not run on hardware
//
#include <hip/hip_runtime.h>
#include <hip/hip_bf16.h>

typedef __attribute__((ext_vector_type(16))) _Float16 v16h;
typedef __attribute__((ext_vector_type(8)))  _Float16 v8h;
typedef __attribute__((ext_vector_type(8)))  float    v8f;

#define BB    4
#define CC    64
#define HH    192
#define WW    192
#define PS    8
#define STR   4
#define PADV  2
#define NPH   48
#define NPW   48
#define NPAT  (NPH * NPW)          // 2304 per image
#define TT    64
#define SCALE 0.25f
#define OUTN  (BB * CC * HH * WW)  // 9437184

// ---------------- WMMA helpers (gfx1250, wave32) ----------------

__device__ __forceinline__ v8f wmma_f16(v16h a, v16h b, v8f c) {
    return __builtin_amdgcn_wmma_f32_16x16x32_f16(
        /*neg_a=*/false, a, /*neg_b=*/false, b,
        /*c_mod=*/(short)0, c, /*reuse_a=*/false, /*reuse_b=*/false);
}

// A fragment 16x32 from row-major src (row stride `stride` elems), rows row0..row0+15, K cols k0..k0+31.
// ISA 16-bit A layout: lane l: M=l%16, g=l/16; VGPR0-3: K=k0+g*8+0..7 ; VGPR4-7: K=k0+16+g*8+0..7.
__device__ __forceinline__ v16h load_a(const _Float16* src, int row0, int stride, int k0) {
    int l = threadIdx.x & 31, m = l & 15, g = l >> 4;
    const _Float16* r = src + (row0 + m) * stride + k0 + g * 8;
    v8h lo = *(const v8h*)(r);
    v8h hi = *(const v8h*)(r + 16);
    v16h a;
#pragma unroll
    for (int i = 0; i < 8; ++i) { a[i] = lo[i]; a[8 + i] = hi[i]; }
    return a;
}

// A fragment with only K=0..15 valid (head-dim 16), src row stride 16; upper K half zeroed.
__device__ __forceinline__ v16h load_a_hd(const _Float16* src, int row0) {
    int l = threadIdx.x & 31, m = l & 15, g = l >> 4;
    v8h lo = *(const v8h*)(src + (row0 + m) * 16 + g * 8);
    v16h a = {};
#pragma unroll
    for (int i = 0; i < 8; ++i) a[i] = lo[i];
    return a;
}

// B fragment 32x16 where B[k][n] = src[(n0+n)*stride + k] (i.e. A @ src^T form).
// B layout (matches A): lane n=l%16, g=l/16; elems 0-7: K=k0+g*8+0..7 ; elems 8-15: K=k0+16+g*8+0..7.
__device__ __forceinline__ v16h load_b_t(const _Float16* src, int n0, int stride, int k0) {
    int l = threadIdx.x & 31, n = l & 15, g = l >> 4;
    const _Float16* r = src + (n0 + n) * stride + k0 + g * 8;
    v8h lo = *(const v8h*)(r);
    v8h hi = *(const v8h*)(r + 16);
    v16h b;
#pragma unroll
    for (int i = 0; i < 8; ++i) { b[i] = lo[i]; b[8 + i] = hi[i]; }
    return b;
}

// B fragment with only K=0..15 valid (head-dim 16): src row stride 16; upper K half (elems 8-15) zeroed.
__device__ __forceinline__ v16h load_b_hd(const _Float16* src, int n0) {
    int l = threadIdx.x & 31, n = l & 15, g = l >> 4;
    v16h b = {};
    v8h lo = *(const v8h*)(src + (n0 + n) * 16 + g * 8);
#pragma unroll
    for (int i = 0; i < 8; ++i) b[i] = lo[i];
    return b;
}

// ---------------- Kernels ----------------

__global__ void zero_out_kernel(float* out) {
    long i = (long)blockIdx.x * 256 + threadIdx.x;
    if (i < OUTN) out[i] = 0.0f;
}

__global__ void wconv_kernel(const float* wq, const float* wk, const float* wv,
                             const float* wp, _Float16* dst) {
    int i = blockIdx.x * 256 + threadIdx.x;
    if (i < 4096) {
        dst[i]         = (_Float16)wq[i];
        dst[4096 + i]  = (_Float16)wk[i];
        dst[8192 + i]  = (_Float16)wv[i];
        dst[12288 + i] = (_Float16)wp[i];
    }
}

// One workgroup (128 threads = 4 waves) per patch. Fully fused attention.
__global__ void __launch_bounds__(128)
patch_attn_kernel(const float* __restrict__ x, const _Float16* __restrict__ wts,
                  float* __restrict__ out) {
    // NOTE: due to the reference's reshape, token index = channel, feature index = spatial.
    __shared__ _Float16 s_tok[TT * 64];      // [token(=channel)][spatial]  8KB
    __shared__ _Float16 s_Q[4][TT * 16];     // [head][token][d], pre-scaled 8KB
    __shared__ _Float16 s_K[4][TT * 16];     // 8KB
    __shared__ _Float16 s_Vt[64 * TT];       // V transposed: [feat][token]  8KB
    __shared__ _Float16 s_P[4 * 16 * 64];    // per-wave softmax stripe      8KB
    __shared__ _Float16 s_O[TT * 64];        // attention output [t][feat]   8KB

    const int patch = blockIdx.x;
    const int bi = patch / NPAT;
    const int pr = patch % NPAT;
    const int ph = pr / NPW, pw = pr % NPW;
    const int h0 = ph * STR - PADV, w0 = pw * STR - PADV;

    const _Float16* wq = wts;
    const _Float16* wk = wts + 4096;
    const _Float16* wvv = wts + 8192;
    const _Float16* wp = wts + 12288;

    const int wid  = threadIdx.x >> 5;
    const int lane = threadIdx.x & 31;
    const int g = lane >> 4, n = lane & 15;

    // ---- Phase 0: gather padded patch into token matrix (token = channel) ----
    for (int r = threadIdx.x; r < 512; r += 128) {
        const int ch = r >> 3, i = r & 7;
        const int hh = h0 + i;
        const float* src = x + (((long)bi * CC + ch) * HH + hh) * WW;
        const bool hv = (hh >= 0 && hh < HH);
#pragma unroll
        for (int j = 0; j < 8; ++j) {
            const int ww = w0 + j;
            float v = (hv && ww >= 0 && ww < WW) ? src[ww] : 0.0f;
            s_tok[ch * 64 + i * 8 + j] = (_Float16)v;
        }
    }
    __syncthreads();

    // ---- Phase 1: Q = tok @ Wq^T (scaled), K = tok @ Wk^T, V = tok @ Wv^T ----
    const int m0 = wid * 16;  // this wave's token stripe
#pragma unroll
    for (int nt = 0; nt < 4; ++nt) {
        v8f q = {}, k = {}, v = {};
#pragma unroll
        for (int ks = 0; ks < 64; ks += 32) {
            v16h a = load_a(s_tok, m0, 64, ks);
            q = wmma_f16(a, load_b_t(wq, nt * 16, 64, ks), q);
            k = wmma_f16(a, load_b_t(wk, nt * 16, 64, ks), k);
            v = wmma_f16(a, load_b_t(wvv, nt * 16, 64, ks), v);
        }
#pragma unroll
        for (int r = 0; r < 8; ++r) {
            const int t = m0 + r + 8 * g;      // token row
            s_Q[nt][t * 16 + n] = (_Float16)(q[r] * SCALE);  // head = nt, d = n
            s_K[nt][t * 16 + n] = (_Float16)k[r];
            s_Vt[(nt * 16 + n) * 64 + t] = (_Float16)v[r];   // transposed store
        }
    }
    __syncthreads();

    // ---- Phase 2: per-head attention, one head per wave ----
    const int hd = wid;
    _Float16* Ph = s_P + wid * (16 * 64);
#pragma unroll 1
    for (int mt = 0; mt < 4; ++mt) {
        v8f sacc[4];
#pragma unroll
        for (int nt = 0; nt < 4; ++nt) {
            v16h a = load_a_hd(s_Q[hd], mt * 16);
            v16h b = load_b_hd(s_K[hd], nt * 16);
            v8f z = {};
            sacc[nt] = wmma_f16(a, b, z);
        }
        // softmax over each row (length 64 spread over 16 lanes x 4 tiles)
#pragma unroll
        for (int r = 0; r < 8; ++r) {
            float mx = fmaxf(fmaxf(sacc[0][r], sacc[1][r]), fmaxf(sacc[2][r], sacc[3][r]));
            mx = fmaxf(mx, __shfl_xor(mx, 8, 32));
            mx = fmaxf(mx, __shfl_xor(mx, 4, 32));
            mx = fmaxf(mx, __shfl_xor(mx, 2, 32));
            mx = fmaxf(mx, __shfl_xor(mx, 1, 32));
            float e0 = __expf(sacc[0][r] - mx);
            float e1 = __expf(sacc[1][r] - mx);
            float e2 = __expf(sacc[2][r] - mx);
            float e3 = __expf(sacc[3][r] - mx);
            float sum = e0 + e1 + e2 + e3;
            sum += __shfl_xor(sum, 8, 32);
            sum += __shfl_xor(sum, 4, 32);
            sum += __shfl_xor(sum, 2, 32);
            sum += __shfl_xor(sum, 1, 32);
            const float inv = 1.0f / sum;
            const int row = r + 8 * g;
            Ph[row * 64 + 0  + n] = (_Float16)(e0 * inv);
            Ph[row * 64 + 16 + n] = (_Float16)(e1 * inv);
            Ph[row * 64 + 32 + n] = (_Float16)(e2 * inv);
            Ph[row * 64 + 48 + n] = (_Float16)(e3 * inv);
        }
        // O stripe = P @ V_head   (same-wave LDS RAW: in-order via DScnt)
        v8f oacc = {};
#pragma unroll
        for (int ks = 0; ks < 64; ks += 32) {
            v16h a = load_a(Ph, 0, 64, ks);
            v16h b = load_b_t(s_Vt, hd * 16, 64, ks);   // B[k][n] = V[k][16h+n]
            oacc = wmma_f16(a, b, oacc);
        }
#pragma unroll
        for (int r = 0; r < 8; ++r) {
            const int t = mt * 16 + r + 8 * g;
            s_O[t * 64 + hd * 16 + n] = (_Float16)oacc[r];
        }
    }
    __syncthreads();

    // ---- Phase 3: out = O @ Wp^T, scatter-add into folded output ----
    const int mt = wid;
#pragma unroll
    for (int nt = 0; nt < 4; ++nt) {
        v8f acc = {};
#pragma unroll
        for (int ks = 0; ks < 64; ks += 32) {
            v16h a = load_a(s_O, mt * 16, 64, ks);
            v16h b = load_b_t(wp, nt * 16, 64, ks);
            acc = wmma_f16(a, b, acc);
        }
#pragma unroll
        for (int r = 0; r < 8; ++r) {
            const int t = mt * 16 + r + 8 * g;  // output channel (token index)
            const int c = nt * 16 + n;          // spatial position within patch
            const int i = c >> 3, j = c & 7;
            const int hh = h0 + i, ww = w0 + j;
            if (hh >= 0 && hh < HH && ww >= 0 && ww < WW) {
                atomicAdd(out + (((long)bi * CC + t) * HH + hh) * WW + ww, acc[r]);
            }
        }
    }
}

__global__ void finalize_kernel(float* out) {
    long i = (long)blockIdx.x * 256 + threadIdx.x;
    if (i >= OUTN) return;
    const int w = (int)(i % WW);
    const int h = (int)((i / WW) % HH);
    int lh = (h - 2) >> 2; if (lh < 0) lh = 0;
    int hh = (h + 2) >> 2; if (hh > NPH - 1) hh = NPH - 1;
    int lw = (w - 2) >> 2; if (lw < 0) lw = 0;
    int hw = (w + 2) >> 2; if (hw > NPW - 1) hw = NPW - 1;
    const float cnt = (float)((hh - lh + 1) * (hw - lw + 1));
    out[i] = out[i] / (cnt + 1e-6f);
}

// ---------------- Launch ----------------

extern "C" void kernel_launch(void* const* d_in, const int* in_sizes, int n_in,
                              void* d_out, int out_size, void* d_ws, size_t ws_size,
                              hipStream_t stream) {
    (void)in_sizes; (void)n_in; (void)out_size; (void)ws_size;
    const float* x  = (const float*)d_in[0];
    const float* Wq = (const float*)d_in[1];
    const float* Wk = (const float*)d_in[2];
    const float* Wv = (const float*)d_in[3];
    const float* Wp = (const float*)d_in[4];
    float* out = (float*)d_out;
    _Float16* wh = (_Float16*)d_ws;   // 16384 f16 = 32KB of workspace

    const int nblk = (OUTN + 255) / 256;
    zero_out_kernel<<<nblk, 256, 0, stream>>>(out);
    wconv_kernel<<<16, 256, 0, stream>>>(Wq, Wk, Wv, Wp, wh);
    patch_attn_kernel<<<BB * NPAT, 128, 0, stream>>>(x, wh, out);
    finalize_kernel<<<nblk, 256, 0, stream>>>(out);
}
